// VFINet_50122268345060
// MI455X (gfx1250) — matrix operand order repaired and checked by reading, not compiled, not run on hardware
//
#include <hip/hip_runtime.h>
#include <hip/hip_bf16.h>

typedef __bf16 bf16;
typedef __attribute__((ext_vector_type(16))) __bf16 v16bf;
typedef __attribute__((ext_vector_type(8)))  __bf16 v8bf;
typedef __attribute__((ext_vector_type(8)))  float  v8f;

#define PI_F 3.14159265358979323846f

static constexpr int Bn = 4;
static constexpr int H1 = 256, W1 = 448, HW1 = H1 * W1;
static constexpr int H2 = 128, W2 = 224, HW2 = H2 * W2;
static constexpr int H4 = 64,  W4 = 112, HW4 = H4 * W4;

// ------------------------------------------------------------ TDM support

#if defined(__has_builtin)
#if __has_builtin(__builtin_amdgcn_tensor_load_to_lds) && __has_builtin(__builtin_amdgcn_s_wait_tensorcnt)
#define HAVE_TDM 1
#endif
#endif

#ifdef HAVE_TDM
typedef __attribute__((ext_vector_type(4))) unsigned int ui32x4;
typedef __attribute__((ext_vector_type(8))) int i32x8;
typedef __attribute__((ext_vector_type(4))) int i32x4;

// Issue a TDM 2D tile load: rows x 32 bf16 elements, global row stride
// kpad_elems, packed contiguously into LDS at lds_off.  D# packing per
// CDNA5 ISA (group0 128b, group1 256b, groups 2/3 zero).
__device__ __forceinline__ void tdm_load_weights(const bf16* gsrc, unsigned lds_off,
                                                 int rows, int kpad_elems) {
    unsigned long long ga = (unsigned long long)(uintptr_t)gsrc;
    ui32x4 g0;
    g0.x = 1u;                                               // count=1, user mode
    g0.y = lds_off;                                          // lds_addr (bytes)
    g0.z = (unsigned)(ga & 0xFFFFFFFFull);                   // global_addr[31:0]
    g0.w = (unsigned)((ga >> 32) & 0x01FFFFFFull) | (2u << 30); // [56:32] | type=2
    i32x8 g1;
    g1[0] = 1 << 16;                                         // data_size = 2 bytes
    g1[1] = (kpad_elems & 0xFFFF) << 16;                     // tensor_dim0[15:0]
    g1[2] = ((kpad_elems >> 16) & 0xFFFF) | ((rows & 0xFFFF) << 16); // dim0 hi | dim1 lo
    g1[3] = ((rows >> 16) & 0xFFFF) | (32 << 16);            // dim1 hi | tile_dim0=32
    g1[4] = rows & 0xFFFF;                                   // tile_dim1=rows, tile_dim2=0
    g1[5] = kpad_elems;                                      // tensor_dim0_stride[31:0]
    g1[6] = 0;
    g1[7] = 0;
    i32x4 z4 = {0, 0, 0, 0};
#if __clang_major__ >= 23
    i32x8 z8 = {0, 0, 0, 0, 0, 0, 0, 0};
    __builtin_amdgcn_tensor_load_to_lds(g0, g1, z4, z4, z8, 0);
#else
    __builtin_amdgcn_tensor_load_to_lds(g0, g1, z4, z4, 0);
#endif
}
#endif // HAVE_TDM

// ---------------------------------------------------------------- utilities

__global__ void fill_zero_kernel(float* __restrict__ p, size_t n) {
    size_t i = (size_t)blockIdx.x * blockDim.x + threadIdx.x;
    size_t stride = (size_t)gridDim.x * blockDim.x;
    for (; i < n; i += stride) p[i] = 0.0f;
}

// re-layout weights OIHW(f32) -> [O][Kpad](bf16) with K = (kh*3+kw)*Cin + c,
// zero-padded to Kpad (multiple of 32) so fragment loads never need tail checks
__global__ void wgt_relayout_kernel(const float* __restrict__ src, bf16* __restrict__ dst,
                                    int Cin, int Cout, int Kpad) {
    int i = blockIdx.x * blockDim.x + threadIdx.x;
    if (i >= Cout * Kpad) return;
    int o = i / Kpad, k = i % Kpad;
    float v = 0.f;
    if (k < Cin * 9) {
        int r = k / Cin, c = k - r * Cin;      // r = kh*3+kw
        v = src[(o * Cin + c) * 9 + r];
    }
    dst[i] = (bf16)v;
}

// mean over concat([img0,img1],1), per-batch; one block per batch element
__global__ void mean_kernel(const float* __restrict__ img0, const float* __restrict__ img1,
                            float* __restrict__ mean) {
    __shared__ float red[256];
    int b = blockIdx.x;
    const int N = 3 * HW1;
    const float* p0 = img0 + (size_t)b * N;
    const float* p1 = img1 + (size_t)b * N;
    float s = 0.f;
    for (int i = threadIdx.x; i < N; i += blockDim.x) s += p0[i] + p1[i];
    red[threadIdx.x] = s;
    __syncthreads();
    for (int off = 128; off > 0; off >>= 1) {
        if ((int)threadIdx.x < off) red[threadIdx.x] += red[threadIdx.x + off];
        __syncthreads();
    }
    if (threadIdx.x == 0) mean[b] = red[0] / (float)(6 * HW1);
}

__global__ void sub_mean_kernel(const float* __restrict__ img, const float* __restrict__ mean,
                                float* __restrict__ outf, bf16* __restrict__ outb) {
    int i = blockIdx.x * blockDim.x + threadIdx.x;
    const int per_b = 3 * HW1;
    if (i >= Bn * per_b) return;
    int b = i / per_b;
    float v = img[i] - mean[b];
    outf[i] = v;
    outb[i] = (bf16)v;
}

__global__ void inter_flow_kernel(const float* __restrict__ flo, const float* __restrict__ sigma,
                                  float* __restrict__ out, float t) {
    int i = blockIdx.x * blockDim.x + threadIdx.x;
    if (i >= Bn * HW1) return;
    int b = i / HW1, p = i % HW1;
    float fx = flo[((size_t)b * 2 + 0) * HW1 + p];
    float fy = flo[((size_t)b * 2 + 1) * HW1 + p];
    float sg = sigma[(size_t)b * HW1 + p];
    float alpha = atan2f(fy, fx);
    float beta  = asinf(sg);
    float d     = sqrtf(fx * fx + fy * fy);
    bool  mask  = fabsf(sg) < 0.01f;
    float R     = mask ? 0.f : d / (2.f * sg);
    float th0 = alpha + PI_F * 0.5f + beta;
    float th1 = alpha + PI_F * 0.5f - beta;
    float tht = th0 + (th1 - th0) * t;
    float ftx = mask ? t * fx : R * (cosf(tht) - cosf(th0));
    float fty = mask ? t * fy : R * (sinf(tht) - sinf(th0));
    out[((size_t)b * 2 + 0) * HW1 + p] = ftx;
    out[((size_t)b * 2 + 1) * HW1 + p] = fty;
}

__global__ void flow_resize_kernel(const float* __restrict__ flo, float* __restrict__ out,
                                   int Hi, int Wi, int Ho, int Wo) {
    int i = blockIdx.x * blockDim.x + threadIdx.x;
    int HWo = Ho * Wo;
    if (i >= Bn * 2 * HWo) return;
    int p  = i % HWo;
    int ch = (i / HWo) & 1;
    int b  = i / (2 * HWo);
    int oy = p / Wo, ox = p % Wo;
    float sy = (float)Hi / (float)Ho, sx = (float)Wi / (float)Wo;
    float fy = (oy + 0.5f) * sy - 0.5f;
    float fx = (ox + 0.5f) * sx - 0.5f;
    float y0f = floorf(fy), x0f = floorf(fx);
    float dy = fy - y0f, dx = fx - x0f;
    int y0 = (int)y0f, x0 = (int)x0f;
    int y0c = min(max(y0, 0), Hi - 1), y1c = min(max(y0 + 1, 0), Hi - 1);
    int x0c = min(max(x0, 0), Wi - 1), x1c = min(max(x0 + 1, 0), Wi - 1);
    const float* src = flo + ((size_t)b * 2 + ch) * Hi * Wi;
    float v = (1.f - dy) * ((1.f - dx) * src[y0c * Wi + x0c] + dx * src[y0c * Wi + x1c])
            +         dy * ((1.f - dx) * src[y1c * Wi + x0c] + dx * src[y1c * Wi + x1c]);
    float scale = (ch == 0) ? (float)Wo / (float)Wi : (float)Ho / (float)Hi;
    out[i] = v * scale;
}

// ------------------------------------------------------------ conv (WMMA)
//
// Implicit-GEMM conv3x3 + bias + PReLU, bf16 in / f32 acc / bf16 out.
// Block tile: COUT x 64 pixels. Each wave owns 16 output channels and 4
// accumulators -> one A fragment feeds 4x v_wmma_f32_16x16x32_bf16 per K-step.
//
// Weight slab (COUT x 32 bf16, row stride KPAD) is DMA'd to LDS by the
// Tensor Data Mover once per K-step (wave 0 issues, waits TENSORcnt), so
// each block loads every weight exactly once instead of per-wave b128 pairs.
//
// Fragment layouts (ISA 16-bit A/B VGPR maps) reduce to contiguous K runs:
//   A elem i -> K = k0 + 8*half + i + (i>=8 ? 8 : 0)   => 2x 16B LDS reads
//   B elem i -> K = k0 + 8*half + i + (i>=8 ? 8 : 0)   => 2x 16B LDS reads
// LDS B-tile is stored K-contiguous per pixel: Bt[n][k], stride 40
// (80B: 16B-aligned, bank-spread).
template <int CIN, int COUT, int HIN, int WIN, int STRIDE>
__global__ __launch_bounds__(2 * COUT)
void conv_wmma_kernel(const bf16* __restrict__ in, const bf16* __restrict__ wgt,
                      const float* __restrict__ bias, const float* __restrict__ alpha,
                      bf16* __restrict__ out) {
    constexpr int HOUT  = HIN / STRIDE, WOUT = WIN / STRIDE;
    constexpr int HWIN  = HIN * WIN,    HWOUT = HOUT * WOUT;
    constexpr int KDIM  = CIN * 9;
    constexpr int KPAD  = (KDIM + 31) & ~31;
    constexpr int NT    = 2 * COUT;          // threads per block
    constexpr int BTS   = 40;                // LDS row stride (bf16)
    constexpr bool ALIGNED = (CIN % 32 == 0);

    const int b     = blockIdx.z;
    const int ptile = blockIdx.x * 64;
    const int lane  = threadIdx.x & 31;
    const int wave  = threadIdx.x >> 5;
    const int half  = lane >> 4;
    const int ln    = lane & 15;

    __shared__ bf16 Bt[64 * BTS];
#ifdef HAVE_TDM
    __shared__ bf16 Wlds[COUT * 32];
#endif

    const bf16* inb  = in + (size_t)b * CIN * HWIN;
    const bf16* wrow = wgt + (size_t)(wave * 16 + ln) * KPAD + 8 * half;
    (void)wrow;

    v8f acc[4] = {};

    for (int k0 = 0; k0 < KDIM; k0 += 32) {
#ifdef HAVE_TDM
        // ---- TDM: DMA this K-step's COUTx32 weight slab into LDS (wave 0)
        if (threadIdx.x < 32) {
            tdm_load_weights(wgt + k0, (unsigned)(uintptr_t)Wlds, COUT, KPAD);
            __builtin_amdgcn_s_wait_tensorcnt(0);
        }
#endif
        // uniform (kh,kw) per K-step when CIN is a multiple of 32
        const int r_uni  = k0 / CIN;
        const int kh_uni = r_uni / 3, kw_uni = r_uni - kh_uni * 3;

        // ---- stage 32(K) x 64(N) im2col tile, K-contiguous per pixel
        for (int idx = threadIdx.x; idx < 2048; idx += NT) {
            int n  = idx & 63;
            int kk = idx >> 6;
            int k  = k0 + kk;
            bf16 v = (bf16)0.f;
            if (ALIGNED || k < KDIM) {
                int c, kh, kw;
                if constexpr (ALIGNED) {
                    c = kk; kh = kh_uni; kw = kw_uni;
                } else {
                    int r = k / CIN;                 // compile-time magic div
                    c = k - r * CIN;
                    kh = r / 3; kw = r - kh * 3;
                }
                int p  = ptile + n;
                int oh = p / WOUT;                   // compile-time magic div
                int ow = p - oh * WOUT;
                int ih = oh * STRIDE + kh - 1;
                int iw = ow * STRIDE + kw - 1;
                if ((unsigned)ih < (unsigned)HIN && (unsigned)iw < (unsigned)WIN) {
                    const bf16* src = inb + (size_t)c * HWIN + ih * WIN + iw;
                    v = *src;
                    __builtin_prefetch(src + 32 * HWIN, 0, 1);  // next K-step line
                }
            }
            Bt[n * BTS + kk] = v;
        }
        __syncthreads();

        // ---- A fragment: two contiguous 16B reads (LDS slab via TDM, else global)
#ifdef HAVE_TDM
        const bf16* wr = &Wlds[(wave * 16 + ln) * 32 + 8 * half];
        v8bf alo = *(const v8bf*)(wr);
        v8bf ahi = *(const v8bf*)(wr + 16);
#else
        v8bf alo = *(const v8bf*)(wrow + k0);
        v8bf ahi = *(const v8bf*)(wrow + k0 + 16);
#endif
        v16bf afrag = __builtin_shufflevector(alo, ahi,
                          0, 1, 2, 3, 4, 5, 6, 7, 8, 9, 10, 11, 12, 13, 14, 15);

        // ---- 4 WMMAs reuse the A fragment across 4 pixel sub-tiles
        #pragma unroll
        for (int nt = 0; nt < 4; ++nt) {
            const bf16* brow = &Bt[(nt * 16 + ln) * BTS + 8 * half];
            v8bf blo = *(const v8bf*)(brow);
            v8bf bhi = *(const v8bf*)(brow + 16);
            v16bf bfrag = __builtin_shufflevector(blo, bhi,
                              0, 1, 2, 3, 4, 5, 6, 7, 8, 9, 10, 11, 12, 13, 14, 15);
            acc[nt] = __builtin_amdgcn_wmma_f32_16x16x32_bf16(false, afrag, false, bfrag,
                                                              (short)0, acc[nt], false, false);
        }
        __syncthreads();
    }

    // ---- bias + PReLU epilogue, bf16 store
    float al = alpha[0];
    #pragma unroll
    for (int nt = 0; nt < 4; ++nt) {
        int p = ptile + nt * 16 + ln;
        #pragma unroll
        for (int r = 0; r < 8; ++r) {
            int o = wave * 16 + r + 8 * half;
            float v = acc[nt][r] + bias[o];
            v = v > 0.f ? v : al * v;
            out[((size_t)b * COUT + o) * HWOUT + p] = (bf16)v;
        }
    }
}

// ------------------------------------------------------------ forward warp

template <typename T>
__global__ void warp_scatter_kernel(const T* __restrict__ feat, const float* __restrict__ flow,
                                    float* __restrict__ num, float* __restrict__ den,
                                    int C, int Ctot, int c_off, int Hh, int Ww) {
    int HWl = Hh * Ww;
    int i = blockIdx.x * blockDim.x + threadIdx.x;
    if (i >= Bn * HWl) return;
    int p = i % HWl, b = i / HWl;
    float fx = flow[((size_t)b * 2 + 0) * HWl + p];
    float fy = flow[((size_t)b * 2 + 1) * HWl + p];
    int gy = p / Ww, gx = p % Ww;
    float tx = gx + fx, ty = gy + fy;
    float x0f = floorf(tx), y0f = floorf(ty);
    float dx = tx - x0f, dy = ty - y0f;
    int x0 = (int)x0f, y0 = (int)y0f;

    int   cx[4] = { x0, x0 + 1, x0,     x0 + 1 };
    int   cy[4] = { y0, y0,     y0 + 1, y0 + 1 };
    float cw[4] = { (1.f - dx) * (1.f - dy), dx * (1.f - dy),
                    (1.f - dx) * dy,         dx * dy };

    #pragma unroll
    for (int q = 0; q < 4; ++q) {
        int xi = cx[q], yi = cy[q];
        if (xi >= 0 && xi < Ww && yi >= 0 && yi < Hh) {
            int idx = yi * Ww + xi;
            float w = cw[q];
            atomicAdd(&den[(size_t)b * HWl + idx], w);
            for (int c = 0; c < C; ++c) {
                float fv = (float)feat[((size_t)b * C + c) * HWl + p];
                atomicAdd(&num[((size_t)b * Ctot + c_off + c) * HWl + idx], fv * w);
            }
        }
    }
}

__global__ void warp_norm_kernel(float* __restrict__ num, const float* __restrict__ den,
                                 int C, int Ctot, int c_off, int HWl) {
    int i = blockIdx.x * blockDim.x + threadIdx.x;
    if (i >= Bn * C * HWl) return;
    int p = i % HWl;
    int c = (i / HWl) % C;
    int b = i / (HWl * C);
    float d = den[(size_t)b * HWl + p];
    num[((size_t)b * Ctot + c_off + c) * HWl + p] /= fmaxf(d, 1e-7f);
}

// ---------------------------------------------------------------- launcher

static inline int cdiv(long long a, long long b) { return (int)((a + b - 1) / b); }

extern "C" void kernel_launch(void* const* d_in, const int* in_sizes, int n_in,
                              void* d_out, int out_size, void* d_ws, size_t ws_size,
                              hipStream_t stream) {
    (void)in_sizes; (void)n_in; (void)ws_size;
    const float* img0    = (const float*)d_in[0];
    const float* img1    = (const float*)d_in[1];
    const float* flow01  = (const float*)d_in[2];
    const float* sigma01 = (const float*)d_in[3];
    const float* flow10  = (const float*)d_in[4];
    const float* sigma10 = (const float*)d_in[5];
    const float* wgtf[6], * biasf[6], * alphaf[6];
    for (int l = 0; l < 6; ++l) {
        wgtf[l]   = (const float*)d_in[6 + 3 * l + 0];
        biasf[l]  = (const float*)d_in[6 + 3 * l + 1];
        alphaf[l] = (const float*)d_in[6 + 3 * l + 2];
    }
    float* out = (float*)d_out;

    // ---- workspace carve-up (byte cursor, 256B aligned)
    char* ws = (char*)d_ws;
    size_t cur = 0;
    auto alloc = [&](size_t bytes) -> char* {
        char* p = ws + cur;
        cur = (cur + bytes + 255) & ~(size_t)255;
        return p;
    };
    float* meanb    = (float*)alloc(Bn * sizeof(float));
    float* img0c_f  = (float*)alloc((size_t)Bn * 3 * HW1 * 4);
    float* img1c_f  = (float*)alloc((size_t)Bn * 3 * HW1 * 4);
    bf16*  img0c_b  = (bf16*) alloc((size_t)Bn * 3 * HW1 * 2);
    bf16*  img1c_b  = (bf16*) alloc((size_t)Bn * 3 * HW1 * 2);
    float* flow0t   = (float*)alloc((size_t)Bn * 2 * HW1 * 4);
    float* flow1t   = (float*)alloc((size_t)Bn * 2 * HW1 * 4);
    float* fl0dd    = (float*)alloc((size_t)Bn * 2 * HW2 * 4);
    float* fl1dd    = (float*)alloc((size_t)Bn * 2 * HW2 * 4);
    float* fl0ddd   = (float*)alloc((size_t)Bn * 2 * HW4 * 4);
    float* fl1ddd   = (float*)alloc((size_t)Bn * 2 * HW4 * 4);
    const int CoutL[6] = {32, 32, 64, 64, 96, 96};
    const int CinL [6] = { 3, 32, 32, 64, 64, 96};
    int KpadL[6];
    bf16* wgtb[6];
    for (int l = 0; l < 6; ++l) {
        KpadL[l] = (CinL[l] * 9 + 31) & ~31;
        wgtb[l] = (bf16*)alloc((size_t)CoutL[l] * KpadL[l] * 2);
    }
    bf16* c1[2], * f1[2], * c3[2], * f2[2], * c5[2], * f3[2];
    for (int im = 0; im < 2; ++im) {
        c1[im] = (bf16*)alloc((size_t)Bn * 32 * HW1 * 2);
        f1[im] = (bf16*)alloc((size_t)Bn * 32 * HW1 * 2);
        c3[im] = (bf16*)alloc((size_t)Bn * 64 * HW2 * 2);
        f2[im] = (bf16*)alloc((size_t)Bn * 64 * HW2 * 2);
        c5[im] = (bf16*)alloc((size_t)Bn * 96 * HW4 * 2);
        f3[im] = (bf16*)alloc((size_t)Bn * 96 * HW4 * 2);
    }
    float* den = (float*)alloc((size_t)Bn * HW1 * 4);

    // ---- output region offsets (concat order of the reference)
    size_t o0 = 0;
    size_t o1 = o0 + (size_t)Bn * 6   * HW1;
    size_t o2 = o1 + (size_t)Bn * 64  * HW1;
    size_t o3 = o2 + (size_t)Bn * 128 * HW2;
    size_t o4 = o3 + (size_t)Bn * 192 * HW4;

    const int TB = 256;

    // 1) zero d_out (harness poisons it)
    fill_zero_kernel<<<2048, TB, 0, stream>>>(out, (size_t)out_size);

    // 2) per-batch mean + subtraction
    mean_kernel<<<Bn, TB, 0, stream>>>(img0, img1, meanb);
    sub_mean_kernel<<<cdiv((size_t)Bn * 3 * HW1, TB), TB, 0, stream>>>(img0, meanb, img0c_f, img0c_b);
    sub_mean_kernel<<<cdiv((size_t)Bn * 3 * HW1, TB), TB, 0, stream>>>(img1, meanb, img1c_f, img1c_b);

    // 3) intermediate flows (t = 0.5 for both directions)
    inter_flow_kernel<<<cdiv((size_t)Bn * HW1, TB), TB, 0, stream>>>(flow01, sigma01, flow0t, 0.5f);
    inter_flow_kernel<<<cdiv((size_t)Bn * HW1, TB), TB, 0, stream>>>(flow10, sigma10, flow1t, 0.5f);

    // 4) weights -> bf16, GEMM-K layout, zero-padded to Kpad
    for (int l = 0; l < 6; ++l) {
        int n = CoutL[l] * KpadL[l];
        wgt_relayout_kernel<<<cdiv(n, TB), TB, 0, stream>>>(wgtf[l], wgtb[l],
                                                            CinL[l], CoutL[l], KpadL[l]);
    }

    // 5) feature extractor: WMMA implicit-GEMM convs (per image)
    for (int im = 0; im < 2; ++im) {
        const bf16* in0 = im ? img1c_b : img0c_b;
        conv_wmma_kernel<3,  32, H1, W1, 1><<<dim3(HW1 / 64, 1, Bn), 64,  0, stream>>>(
            in0,    wgtb[0], biasf[0], alphaf[0], c1[im]);
        conv_wmma_kernel<32, 32, H1, W1, 1><<<dim3(HW1 / 64, 1, Bn), 64,  0, stream>>>(
            c1[im], wgtb[1], biasf[1], alphaf[1], f1[im]);
        conv_wmma_kernel<32, 64, H1, W1, 2><<<dim3(HW2 / 64, 1, Bn), 128, 0, stream>>>(
            f1[im], wgtb[2], biasf[2], alphaf[2], c3[im]);
        conv_wmma_kernel<64, 64, H2, W2, 1><<<dim3(HW2 / 64, 1, Bn), 128, 0, stream>>>(
            c3[im], wgtb[3], biasf[3], alphaf[3], f2[im]);
        conv_wmma_kernel<64, 96, H2, W2, 2><<<dim3(HW4 / 64, 1, Bn), 192, 0, stream>>>(
            f2[im], wgtb[4], biasf[4], alphaf[4], c5[im]);
        conv_wmma_kernel<96, 96, H4, W4, 1><<<dim3(HW4 / 64, 1, Bn), 192, 0, stream>>>(
            c5[im], wgtb[5], biasf[5], alphaf[5], f3[im]);
    }

    // 6) flow pyramids (level-1 flows equal flow0t/flow1t: identity resize, scale 1)
    flow_resize_kernel<<<cdiv((size_t)Bn * 2 * HW2, TB), TB, 0, stream>>>(flow0t, fl0dd,  H1, W1, H2, W2);
    flow_resize_kernel<<<cdiv((size_t)Bn * 2 * HW2, TB), TB, 0, stream>>>(flow1t, fl1dd,  H1, W1, H2, W2);
    flow_resize_kernel<<<cdiv((size_t)Bn * 2 * HW4, TB), TB, 0, stream>>>(flow0t, fl0ddd, H1, W1, H4, W4);
    flow_resize_kernel<<<cdiv((size_t)Bn * 2 * HW4, TB), TB, 0, stream>>>(flow1t, fl1ddd, H1, W1, H4, W4);

    // 7) forward warps: zero-den -> scatter -> normalize, into d_out slices
    auto warp_f32 = [&](const float* feat, const float* flow, float* dst,
                        int C, int Ctot, int c_off, int Hh, int Ww) {
        int HWl = Hh * Ww;
        fill_zero_kernel<<<cdiv((size_t)Bn * HWl, TB), TB, 0, stream>>>(den, (size_t)Bn * HWl);
        warp_scatter_kernel<float><<<cdiv((size_t)Bn * HWl, TB), TB, 0, stream>>>(
            feat, flow, dst, den, C, Ctot, c_off, Hh, Ww);
        warp_norm_kernel<<<cdiv((size_t)Bn * C * HWl, TB), TB, 0, stream>>>(
            dst, den, C, Ctot, c_off, HWl);
    };
    auto warp_bf16 = [&](const bf16* feat, const float* flow, float* dst,
                         int C, int Ctot, int c_off, int Hh, int Ww) {
        int HWl = Hh * Ww;
        fill_zero_kernel<<<cdiv((size_t)Bn * HWl, TB), TB, 0, stream>>>(den, (size_t)Bn * HWl);
        warp_scatter_kernel<bf16><<<cdiv((size_t)Bn * HWl, TB), TB, 0, stream>>>(
            feat, flow, dst, den, C, Ctot, c_off, Hh, Ww);
        warp_norm_kernel<<<cdiv((size_t)Bn * C * HWl, TB), TB, 0, stream>>>(
            dst, den, C, Ctot, c_off, HWl);
    };

    warp_f32 (img0c_f, flow0t, out + o0, 3,  6,   0,  H1, W1);  // img_0t
    warp_f32 (img1c_f, flow1t, out + o0, 3,  6,   3,  H1, W1);  // img_1t
    warp_bf16(f1[0],   flow0t, out + o1, 32, 64,  0,  H1, W1);  // ft0_1
    warp_bf16(f1[1],   flow1t, out + o1, 32, 64,  32, H1, W1);  // ft1_1
    warp_bf16(f2[0],   fl0dd,  out + o2, 64, 128, 0,  H2, W2);  // ft0_2
    warp_bf16(f2[1],   fl1dd,  out + o2, 64, 128, 64, H2, W2);  // ft1_2
    warp_bf16(f3[0],   fl0ddd, out + o3, 96, 192, 0,  H4, W4);  // ft0_3
    warp_bf16(f3[1],   fl1ddd, out + o3, 96, 192, 96, H4, W4);  // ft1_3
    warp_f32 (sigma01, flow0t, out + o4, 1,  2,   0,  H1, W1);  // sig_0t
    warp_f32 (sigma10, flow1t, out + o4, 1,  2,   1,  H1, W1);  // sig_1t
}
